// ANIInteraction_25769804392
// MI455X (gfx1250) — matrix operand order repaired and checked by reading, not compiled, not run on hardware
//
#include <hip/hip_runtime.h>
#include <hip/hip_bf16.h>

// ---------------- types for WMMA ----------------
typedef __attribute__((ext_vector_type(16))) __bf16 v16bf;
typedef __attribute__((ext_vector_type(8)))  float  v8f;

union Frag32B { uint4 u[2]; v16bf v; };

// ---------------- problem constants ----------------
#define N_ATOMS   131072
#define AEVD      1008
#define K1P       1024        // AEV K padded to multiple of 32
#define N1        256
#define N2        192
#define N3        160
#define OUT_DIM   2
#define NSPEC     7
#define MT        64          // atoms per block (4 WMMA M-tiles)
#define THREADS   256         // 8 wave32s: 2 waves/SIMD -> big VGPR budget, no spills
#define NWAVES    8
#define PERM_LEN  (N_ATOMS + NSPEC * MT)

// ======================================================================
// Species compaction
// ======================================================================
__global__ void k_init(int* counts, int* cursors, int* perm) {
    int i = blockIdx.x * blockDim.x + threadIdx.x;
    if (i < 8)  counts[i]  = 0;
    if (i < 8)  cursors[i] = 0;
    if (i < PERM_LEN) perm[i] = -1;
}

__global__ void k_count(const int* __restrict__ species, int* counts) {
    int i = blockIdx.x * blockDim.x + threadIdx.x;
    if (i < N_ATOMS) atomicAdd(&counts[species[i]], 1);
}

__global__ void k_scan(const int* __restrict__ counts, int* aoff) {
    if (threadIdx.x == 0 && blockIdx.x == 0) {
        int acc = 0;
        aoff[0] = 0;
        for (int e = 0; e < NSPEC; ++e) {
            acc += ((counts[e] + MT - 1) / MT) * MT;   // 64-align each segment
            aoff[e + 1] = acc;
        }
    }
}

__global__ void k_scatter(const int* __restrict__ species,
                          const int* __restrict__ aoff,
                          int* cursors, int* perm) {
    int i = blockIdx.x * blockDim.x + threadIdx.x;
    if (i < N_ATOMS) {
        int s = species[i];
        int pos = aoff[s] + atomicAdd(&cursors[s], 1);
        perm[pos] = i;
    }
}

// ======================================================================
// Weight prep: fp32 [e][Ksrc][N] -> bf16 swizzled WMMA-B layout,
// K zero-padded to Kpad.  dst elem index:
//   (((e*nNt + nt)*nKc + kc)*32 + lane)*16 + j
//   lane<16 : n = nt*16+lane,    K = kc*32 + 0..7 then 16..23
//   lane>=16: n = nt*16+lane-16, K = kc*32 + 8..15 then 24..31
// ======================================================================
__global__ void k_prep(const float* __restrict__ W, __bf16* __restrict__ dst,
                       int Ksrc, int Kpad, int N) {
    int nKc = Kpad >> 5, nNt = N >> 4;
    size_t total = (size_t)NSPEC * nKc * nNt * 512;
    size_t gid = (size_t)blockIdx.x * blockDim.x + threadIdx.x;
    if (gid >= total) return;
    int j    = (int)(gid & 15);
    int lane = (int)((gid >> 4) & 31);
    size_t rest = gid >> 9;
    int kc = (int)(rest % nKc);  rest /= nKc;
    int nt = (int)(rest % nNt);
    int e  = (int)(rest / nNt);
    int k = kc * 32 + ((j >> 3) << 4) + ((lane >> 4) << 3) + (j & 7);
    int n = nt * 16 + (lane & 15);
    float v = (k < Ksrc) ? W[((size_t)e * Ksrc + k) * N + n] : 0.f;
    dst[gid] = (__bf16)v;
}

// ======================================================================
// Fragment loaders
// ======================================================================
// A fragment from LDS, 16-bit A 16x32 layout:
//  lanes 0-15 (M=lane):    K=k0+0..7 then K=k0+16..23
//  lanes 16-31 (M=lane-16):K=k0+8..15 then K=k0+24..31
__device__ inline v16bf lds_frag_a(const __bf16* base, int pitch,
                                   int m0, int k0, int lane) {
    int m = m0 + (lane & 15);
    int k = k0 + ((lane >> 4) << 3);
    const __bf16* p = base + m * pitch + k;
    Frag32B f;
    f.u[0] = *(const uint4*)(p);
    f.u[1] = *(const uint4*)(p + 16);
    return f.v;
}

// B fragment from pre-swizzled global buffer: 32 contiguous bytes per lane.
__device__ inline v16bf glb_frag_b(const __bf16* bsw, int tileIdx, int lane) {
    const uint4* p = (const uint4*)(bsw + ((size_t)tileIdx * 32 + lane) * 16);
    Frag32B f;
    f.u[0] = p[0];
    f.u[1] = p[1];
    return f.v;
}

__device__ inline float celu_f(float x) {
    // alpha = 0.1:  x>0 ? x : 0.1*(exp(10x)-1)
    return x > 0.f ? x : 0.1f * (__expf(x * 10.f) - 1.f);
}

// One bf16 GEMM layer: O[MT x N] = celu(A[MT x Kpad] @ Bsw + bias)
__device__ inline void gemm_layer(const __bf16* A, int Kpad,
                                  const __bf16* Bsw, const float* bias,
                                  int N, __bf16* O) {
    int wave = threadIdx.x >> 5;
    int lane = threadIdx.x & 31;
    int nKc = Kpad >> 5;
    int nNt = N >> 4;
    int nMt = MT >> 4;
    int nTiles = nMt * nNt;
    for (int t = wave; t < nTiles; t += NWAVES) {
        int mt = t % nMt;
        int nt = t / nMt;
        v8f c = {0.f, 0.f, 0.f, 0.f, 0.f, 0.f, 0.f, 0.f};
        for (int kc = 0; kc < nKc; ++kc) {
            if (kc + 4 < nKc)
                __builtin_prefetch((const char*)(Bsw +
                    ((size_t)(nt * nKc + kc + 4) * 32 + lane) * 16), 0, 1);
            v16bf a = lds_frag_a(A, Kpad, mt * 16, kc * 32, lane);
            v16bf b = glb_frag_b(Bsw, nt * nKc + kc, lane);
            c = __builtin_amdgcn_wmma_f32_16x16x32_bf16(
                    false, a, false, b, (short)0, c, false, false);
        }
        // epilogue: C layout -> lane&15 = n, VGPR r = row (+8 for lanes 16-31)
        int n = nt * 16 + (lane & 15);
        float bv = bias[n];
        int mbase = mt * 16 + ((lane >> 4) << 3);
#pragma unroll
        for (int r = 0; r < 8; ++r) {
            float y = celu_f(c[r] + bv);
            O[(mbase + r) * N + n] = (__bf16)y;
        }
    }
}

// ======================================================================
// Main fused MLP kernel: one block = 64 compacted atoms of one species.
// ======================================================================
struct alignas(8) Bf4 { __bf16 a, b, c, d; };

__global__ __launch_bounds__(THREADS) void k_mlp(
        const int* __restrict__ perm, const int* __restrict__ aoff,
        const float* __restrict__ aev,
        const __bf16* __restrict__ w1sw, const float* __restrict__ b1,
        const __bf16* __restrict__ w2sw, const float* __restrict__ b2,
        const __bf16* __restrict__ w3sw, const float* __restrict__ b3,
        const float* __restrict__ wh, const float* __restrict__ bh,
        float* __restrict__ out) {
    __shared__ __bf16 s0[MT * K1P];   // 128 KB : layer-1 input (AEV bf16)
    __shared__ __bf16 s1[MT * N1];    //  32 KB : layer-1 out
    __shared__ __bf16 s2[MT * N2];    //  24 KB : layer-2 out
    __shared__ __bf16 s3[MT * N3];    //  20 KB : layer-3 out
    __shared__ int sidx[MT];

    int tile = blockIdx.x;
    int row0 = tile * MT;
    if (row0 >= aoff[NSPEC]) return;
    int e = 0;
    while (e < NSPEC - 1 && row0 >= aoff[e + 1]) ++e;

    int tid = threadIdx.x;
    if (tid < MT) sidx[tid] = perm[row0 + tid];
    __syncthreads();

    // ---- stage AEV rows -> bf16 LDS (zero for padding rows/cols) ----
    for (int i = tid; i < MT * (AEVD / 4); i += THREADS) {
        int r = i / (AEVD / 4), c = i % (AEVD / 4);
        int a = sidx[r];
        float4 f = (a >= 0) ? ((const float4*)(aev + (size_t)a * AEVD))[c]
                            : make_float4(0.f, 0.f, 0.f, 0.f);
        Bf4 q = {(__bf16)f.x, (__bf16)f.y, (__bf16)f.z, (__bf16)f.w};
        *(Bf4*)(&s0[r * K1P + c * 4]) = q;
    }
    for (int i = tid; i < MT * (K1P - AEVD) / 4; i += THREADS) {
        int r = i / ((K1P - AEVD) / 4), c = AEVD + 4 * (i % ((K1P - AEVD) / 4));
        Bf4 z = {(__bf16)0.f, (__bf16)0.f, (__bf16)0.f, (__bf16)0.f};
        *(Bf4*)(&s0[r * K1P + c]) = z;
    }
    __syncthreads();

    gemm_layer(s0, K1P, w1sw + (size_t)e * K1P * N1, b1 + e * N1, N1, s1);
    __syncthreads();
    gemm_layer(s1, N1, w2sw + (size_t)e * N1 * N2, b2 + e * N2, N2, s2);
    __syncthreads();
    gemm_layer(s2, N2, w3sw + (size_t)e * N2 * N3, b3 + e * N3, N3, s3);
    __syncthreads();

    // ---- head: 160x2 per atom in VALU ----
    if (tid < MT * OUT_DIM) {
        int a = tid >> 1, o = tid & 1;
        int idx = sidx[a];
        if (idx >= 0) {
            const float* W = wh + (size_t)e * N3 * OUT_DIM + o;
            float s = bh[e * OUT_DIM + o];
#pragma unroll 4
            for (int k = 0; k < N3; ++k)
                s += (float)s3[a * N3 + k] * W[k * OUT_DIM];
            out[(size_t)idx * OUT_DIM + o] = s;
        }
    }
}

// ======================================================================
// Host launcher
// ======================================================================
extern "C" void kernel_launch(void* const* d_in, const int* in_sizes, int n_in,
                              void* d_out, int out_size, void* d_ws, size_t ws_size,
                              hipStream_t stream) {
    const int*   species = (const int*)  d_in[0];
    const float* aev     = (const float*)d_in[1];
    const float* W1      = (const float*)d_in[2];
    const float* b1      = (const float*)d_in[3];
    const float* W2      = (const float*)d_in[4];
    const float* b2      = (const float*)d_in[5];
    const float* W3      = (const float*)d_in[6];
    const float* b3      = (const float*)d_in[7];
    const float* Wh      = (const float*)d_in[8];
    const float* bh      = (const float*)d_in[9];
    float* out = (float*)d_out;

    // workspace layout
    int* counts  = (int*)d_ws;          // 8
    int* cursors = counts + 8;          // 8
    int* aoff    = cursors + 8;         // 8
    int* perm    = aoff + 8;            // PERM_LEN
    size_t off = ((size_t)(24 + PERM_LEN) * sizeof(int) + 255) & ~(size_t)255;
    __bf16* w1sw = (__bf16*)((char*)d_ws + off);
    off += (size_t)NSPEC * K1P * N1 * sizeof(__bf16);
    __bf16* w2sw = (__bf16*)((char*)d_ws + off);
    off += (size_t)NSPEC * N1 * N2 * sizeof(__bf16);
    __bf16* w3sw = (__bf16*)((char*)d_ws + off);

    // 1) compaction
    k_init   <<<(PERM_LEN + 255) / 256, 256, 0, stream>>>(counts, cursors, perm);
    k_count  <<<N_ATOMS / 256, 256, 0, stream>>>(species, counts);
    k_scan   <<<1, 1, 0, stream>>>(counts, aoff);
    k_scatter<<<N_ATOMS / 256, 256, 0, stream>>>(species, aoff, cursors, perm);

    // 2) weight conversion + swizzle to WMMA-B layout (bf16)
    {
        size_t t1 = (size_t)NSPEC * K1P * N1;
        size_t t2 = (size_t)NSPEC * N1 * N2;
        size_t t3 = (size_t)NSPEC * N2 * N3;
        k_prep<<<(unsigned)((t1 + 255) / 256), 256, 0, stream>>>(W1, w1sw, AEVD, K1P, N1);
        k_prep<<<(unsigned)((t2 + 255) / 256), 256, 0, stream>>>(W2, w2sw, N1, N1, N2);
        k_prep<<<(unsigned)((t3 + 255) / 256), 256, 0, stream>>>(W3, w3sw, N2, N2, N3);
    }

    // 3) fused per-species MLP (max padded tiles: every segment 64-aligned)
    int maxTiles = N_ATOMS / MT + NSPEC;
    k_mlp<<<maxTiles, THREADS, 0, stream>>>(perm, aoff, aev,
                                            w1sw, b1, w2sw, b2, w3sw, b3,
                                            Wh, bh, out);
}
